// RecurrentEncoder_7155415515248
// MI455X (gfx1250) — compile-verified, matrix operand order checked by reading, not run on hardware
//
#include <hip/hip_runtime.h>
#include <hip/hip_bf16.h>

#define N_NODES 65536
#define T_STEPS 24
#define L_LAYERS 2
#define HD 256
#define NSTAT 5
#define NDYN 11

typedef __attribute__((ext_vector_type(16))) __bf16 v16bf;
typedef __attribute__((ext_vector_type(8)))  __bf16 v8bf;
typedef __attribute__((ext_vector_type(8)))  float  v8f;

#define AS1 __attribute__((address_space(1)))
typedef const __bf16 AS1* gcbf16;
typedef const float  AS1* gcfloat;
typedef const v8bf   AS1* gcv8bf;

__device__ __forceinline__ v8f wmma_bf16(v16bf a, v16bf b, v8f c) {
  // D = A(16x32) * B(32x16) + C, f32 accumulate
  return __builtin_amdgcn_wmma_f32_16x16x32_bf16(false, a, false, b, (short)0, c,
                                                 false, false);
}

// Fragment load from a lane-adjusted base pointer + compile-time element offset.
// base must already include (lane&15)*stride + ((lane>>4)<<3); the two 16B
// chunks sit at +0 and +16 elements (ISA 7.12.2 16-bit striping).
__device__ __forceinline__ v16bf ldfrag_ofs(const __bf16* base, int eofs) {
  v16bf o;
  ((v8bf*)&o)[0] = *(const v8bf*)(base + eofs);
  ((v8bf*)&o)[1] = *(const v8bf*)(base + eofs + 16);
  return o;
}

// Same, but explicitly global address space -> global_load_b128 (LOADcnt only,
// SGPR base + offset), decoupled from DS waits.
__device__ __forceinline__ v16bf ldfrag_g(gcbf16 base, int eofs) {
  v16bf o;
  ((v8bf*)&o)[0] = *(gcv8bf)(base + eofs);
  ((v8bf*)&o)[1] = *(gcv8bf)(base + eofs + 16);
  return o;
}

__device__ __forceinline__ float sigm(float x) { return 1.0f / (1.0f + __expf(-x)); }

// ---------------------------------------------------------------------------
// One 16-unit gate pass. k-outer: per k the 4 A-frags are loaded ONCE (they
// are gate-independent), then two gate-pair sub-steps of {8 VMEM, 8 WMMA}.
// sched_group_barrier pins {DS8, VMEM8, WMMA8, VMEM8, WMMA8} per k.
// ---------------------------------------------------------------------------
__device__ __forceinline__ void gate_pass(const __bf16* __restrict__ abase,
                                          const __bf16* __restrict__ hbase,
                                          gcbf16 wibase, gcbf16 whbase,
                                          gcfloat bias_p,
                                          v8f& c0, v8f& c1,
                                          __bf16* hn0, __bf16* hn1) {
  v8f acc[4][2];
  #pragma unroll
  for (int g = 0; g < 4; ++g) {
    float bv = bias_p[g * HD];
    #pragma unroll
    for (int mt = 0; mt < 2; ++mt)
      #pragma unroll
      for (int r = 0; r < 8; ++r) acc[g][mt][r] = bv;
  }
  __builtin_amdgcn_sched_group_barrier(0x020, 4, 0);  // the 4 bias loads

  #pragma unroll
  for (int k = 0; k < 8; ++k) {
    v16bf a0 = ldfrag_ofs(abase, k * 32);
    v16bf a1 = ldfrag_ofs(abase, 16 * HD + k * 32);
    v16bf h0 = ldfrag_ofs(hbase, k * 32);
    v16bf h1 = ldfrag_ofs(hbase, 16 * HD + k * 32);
    __builtin_amdgcn_sched_group_barrier(0x100, 8, 0);  // 8 DS reads
    #pragma unroll
    for (int gp = 0; gp < 2; ++gp) {
      const int g0 = 2 * gp, g1 = 2 * gp + 1;
      v16bf bi0 = ldfrag_g(wibase, g0 * HD * HD + k * 32);
      v16bf bh0 = ldfrag_g(whbase, g0 * HD * HD + k * 32);
      v16bf bi1 = ldfrag_g(wibase, g1 * HD * HD + k * 32);
      v16bf bh1 = ldfrag_g(whbase, g1 * HD * HD + k * 32);
      acc[g0][0] = wmma_bf16(a0, bi0, acc[g0][0]);
      acc[g0][1] = wmma_bf16(a1, bi0, acc[g0][1]);
      acc[g1][0] = wmma_bf16(a0, bi1, acc[g1][0]);
      acc[g1][1] = wmma_bf16(a1, bi1, acc[g1][1]);
      acc[g0][0] = wmma_bf16(h0, bh0, acc[g0][0]);
      acc[g0][1] = wmma_bf16(h1, bh0, acc[g0][1]);
      acc[g1][0] = wmma_bf16(h0, bh1, acc[g1][0]);
      acc[g1][1] = wmma_bf16(h1, bh1, acc[g1][1]);
      __builtin_amdgcn_sched_group_barrier(0x020, 8, 0);  // 8 VMEM reads
      __builtin_amdgcn_sched_group_barrier(0x008, 8, 0);  // 8 WMMA
    }
  }

  #pragma unroll
  for (int mt = 0; mt < 2; ++mt) {
    v8f& cc = mt ? c1 : c0;
    __bf16* hh = mt ? hn1 : hn0;
    #pragma unroll
    for (int r = 0; r < 8; ++r) {
      float iv = sigm(acc[0][mt][r]);
      float fv = sigm(acc[1][mt][r]);
      float gv = tanhf(acc[2][mt][r]);
      float ov = sigm(acc[3][mt][r]);
      float cn = fv * cc[r] + iv * gv;
      cc[r] = cn;
      hh[r] = (__bf16)(ov * tanhf(cn));
    }
  }
}

// ---------------------------------------------------------------------------
// Prep: fp32 -> bf16 weight conversion + padded W_in + combined bias, into ws.
// ---------------------------------------------------------------------------
__global__ void prep_weights(const float* __restrict__ W_in,
                             const float* __restrict__ W_ih,
                             const float* __restrict__ W_hh,
                             const float* __restrict__ b_ih,
                             const float* __restrict__ b_hh,
                             __bf16* __restrict__ wih_b,
                             __bf16* __restrict__ whh_b,
                             __bf16* __restrict__ win_p,
                             float* __restrict__ bias) {
  int i = blockIdx.x * 256 + threadIdx.x;
  const int tot = L_LAYERS * 4 * HD * HD;  // 524288
  if (i < tot) {
    wih_b[i] = (__bf16)W_ih[i];
    whh_b[i] = (__bf16)W_hh[i];
  }
  if (i < HD * 32) {  // W_in [256][16] -> [256][32] zero-padded K
    int r = i >> 5, c = i & 31;
    win_p[i] = (c < 16) ? (__bf16)W_in[r * 16 + c] : (__bf16)0.f;
  }
  if (i < L_LAYERS * 4 * HD) bias[i] = b_ih[i] + b_hh[i];
}

// ---------------------------------------------------------------------------
// Persistent LSTM: each block owns 32 nodes for all T steps and both layers.
// h kept in LDS (bf16), c in registers (f32, D-tile layout).
// ---------------------------------------------------------------------------
__global__ __launch_bounds__(256, 1)
void lstm_persistent(const float* __restrict__ node_feat,
                     const float* __restrict__ dyn_feat,
                     const __bf16* __restrict__ wih,
                     const __bf16* __restrict__ whh,
                     const __bf16* __restrict__ winp,
                     const float* __restrict__ bias,
                     float* __restrict__ out) {
  __shared__ __bf16 sh_h[L_LAYERS][32][HD];  // hidden state, both layers (32KB)
  __shared__ __bf16 sh_xs[32][HD];           // x @ W_in^T for current t (16KB)
  __shared__ __bf16 sh_x[32][32];            // raw 16 features, K-padded (2KB)

  const int tid  = threadIdx.x;
  const int lane = tid & 31;
  const int wave = tid >> 5;
  const int u0   = wave * 32;            // this wave's hidden-unit slice
  const int base = blockIdx.x * 32;      // node tile base
  const int rl   = lane & 15;            // fragment row/col within tile
  const int kb   = (lane >> 4) << 3;     // per-lane K sub-offset

  for (int i = tid; i < L_LAYERS * 32 * HD; i += 256) (&sh_h[0][0][0])[i] = (__bf16)0.f;
  for (int i = tid; i < 32 * 16; i += 256) sh_x[i >> 4][16 + (i & 15)] = (__bf16)0.f;

  v8f creg[L_LAYERS][2][2];  // cell state, D-tile layout [l][ct][mt]
  #pragma unroll
  for (int l = 0; l < L_LAYERS; ++l)
    #pragma unroll
    for (int ct = 0; ct < 2; ++ct)
      #pragma unroll
      for (int mt = 0; mt < 2; ++mt)
        #pragma unroll
        for (int r = 0; r < 8; ++r) creg[l][ct][mt][r] = 0.f;

  // lane-adjusted fragment base pointers (all later offsets are constants)
  const __bf16* xfrag  = &sh_x[0][0] + rl * 32 + kb;          // A frags of raw x
  const int     mb     = (lane >> 4) << 3;                    // D-tile row base
  __bf16*       xs_st  = &sh_xs[0][0] + mb * HD + u0 + rl;    // xs D stores
  const __bf16* xs_a   = &sh_xs[0][0] + rl * HD + kb;         // xs as A frags

  __syncthreads();

  for (int t = 0; t < T_STEPS; ++t) {
    // Opaque per-iteration copies of the readonly weight pointers. Without
    // this, LICM hoists ALL t-invariant weight fragment loads out of the t
    // loop (legal: args are readonly/noalias) -> ~2K VGPRs of hoisted frags
    // -> mass scratch spilling. The empty asm makes the base loop-variant.
    const __bf16* wih_t  = wih;
    const __bf16* whh_t  = whh;
    const __bf16* winp_t = winp;
    const float*  bias_t = bias;
    asm volatile("" : "+s"(wih_t), "+s"(whh_t), "+s"(winp_t), "+s"(bias_t));
    // Recover global address space (flat_load would tick DScnt and couple
    // weight-load waits with the LDS fragment waits).
    gcbf16  wih_g  = (gcbf16)wih_t;
    gcbf16  whh_g  = (gcbf16)whh_t;
    gcbf16  winp_g = (gcbf16)winp_t;
    gcfloat bias_g = (gcfloat)bias_t;

    // --- build raw input tile [32 nodes][16 feats] ---
    for (int i = tid; i < 512; i += 256) {
      int m = i >> 4, f = i & 15;
      int node = base + m;
      float v = (f < NSTAT) ? node_feat[node * NSTAT + f]
                            : dyn_feat[(node * NDYN + (f - NSTAT)) * T_STEPS + t];
      sh_x[m][f] = (__bf16)v;
    }
    __syncthreads();

    // --- xs = x @ W_in^T (one K-step WMMA per 16x16 tile) ---
    __builtin_amdgcn_sched_barrier(0);
    {
      #pragma unroll
      for (int ct = 0; ct < 2; ++ct) {
        v16bf b = ldfrag_g(winp_g + (u0 + ct * 16 + rl) * 32 + kb, 0);
        #pragma unroll
        for (int mt = 0; mt < 2; ++mt) {
          v8f z;
          #pragma unroll
          for (int r = 0; r < 8; ++r) z[r] = 0.f;
          v16bf a = ldfrag_ofs(xfrag, mt * 16 * 32);
          v8f d = wmma_bf16(a, b, z);
          #pragma unroll
          for (int r = 0; r < 8; ++r)
            xs_st[(mt * 16 + r) * HD + ct * 16] = (__bf16)d[r];
        }
      }
    }
    __builtin_amdgcn_sched_barrier(0);
    __syncthreads();

    // --- two LSTM layers ---
    #pragma unroll
    for (int l = 0; l < L_LAYERS; ++l) {
      const __bf16* inp_a = (l == 0) ? xs_a : (&sh_h[0][0][0] + rl * HD + kb);
      const __bf16* h_a   = &sh_h[l][0][0] + rl * HD + kb;
      gcbf16  wi_b = wih_g + l * 4 * HD * HD + (u0 + rl) * HD + kb;
      gcbf16  wh_b = whh_g + l * 4 * HD * HD + (u0 + rl) * HD + kb;
      gcfloat bs_p = bias_g + l * 4 * HD + u0 + rl;

      __bf16 hn[2][2][8];  // [ct][mt][r]
      gate_pass(inp_a, h_a, wi_b, wh_b, bs_p,
                creg[l][0][0], creg[l][0][1], hn[0][0], hn[0][1]);
      __builtin_amdgcn_sched_barrier(0);
      gate_pass(inp_a, h_a, wi_b + 16 * HD, wh_b + 16 * HD, bs_p + 16,
                creg[l][1][0], creg[l][1][1], hn[1][0], hn[1][1]);
      __builtin_amdgcn_sched_barrier(0);

      __syncthreads();  // everyone done reading old h[l] / inp
      {
        __bf16* hst = &sh_h[l][0][0] + mb * HD + u0 + rl;
        #pragma unroll
        for (int ct = 0; ct < 2; ++ct)
          #pragma unroll
          for (int mt = 0; mt < 2; ++mt)
            #pragma unroll
            for (int r = 0; r < 8; ++r)
              hst[(mt * 16 + r) * HD + ct * 16] = hn[ct][mt][r];
      }
      __syncthreads();  // new h[l] visible before next layer/step
    }
  }

  // --- outputs: out[0][l][n][h] = h, out[1][l][n][h] = c ---
  for (int i = tid; i < L_LAYERS * 32 * HD; i += 256) {
    int l = i / (32 * HD);
    int rem = i - l * 32 * HD;
    int m = rem >> 8, u = rem & 255;
    out[(size_t)l * N_NODES * HD + (size_t)(base + m) * HD + u] = (float)sh_h[l][m][u];
  }
  {
    #pragma unroll
    for (int l = 0; l < L_LAYERS; ++l) {
      float* cp = out + (size_t)(L_LAYERS + l) * N_NODES * HD +
                  (size_t)(base + mb) * HD + u0 + rl;
      #pragma unroll
      for (int ct = 0; ct < 2; ++ct)
        #pragma unroll
        for (int mt = 0; mt < 2; ++mt)
          #pragma unroll
          for (int r = 0; r < 8; ++r)
            cp[(mt * 16 + r) * HD + ct * 16] = creg[l][ct][mt][r];
    }
  }
}

extern "C" void kernel_launch(void* const* d_in, const int* in_sizes, int n_in,
                              void* d_out, int out_size, void* d_ws, size_t ws_size,
                              hipStream_t stream) {
  const float* node_feat = (const float*)d_in[0];  // [N,5]
  const float* dyn_feat  = (const float*)d_in[1];  // [N,11,T]
  const float* W_in      = (const float*)d_in[2];  // [256,16]
  const float* W_ih      = (const float*)d_in[3];  // [2,1024,256]
  const float* W_hh      = (const float*)d_in[4];  // [2,1024,256]
  const float* b_ih      = (const float*)d_in[5];  // [2,1024]
  const float* b_hh      = (const float*)d_in[6];  // [2,1024]
  float* out = (float*)d_out;

  // ws layout: wih_bf16 | whh_bf16 | win_pad_bf16 | bias_f32  (~2.1 MB)
  __bf16* wih_b = (__bf16*)d_ws;
  __bf16* whh_b = wih_b + (size_t)L_LAYERS * 4 * HD * HD;
  __bf16* win_p = whh_b + (size_t)L_LAYERS * 4 * HD * HD;
  float*  bias  = (float*)(win_p + (size_t)HD * 32);

  prep_weights<<<(L_LAYERS * 4 * HD * HD + 255) / 256, 256, 0, stream>>>(
      W_in, W_ih, W_hh, b_ih, b_hh, wih_b, whh_b, win_p, bias);

  lstm_persistent<<<N_NODES / 32, 256, 0, stream>>>(
      node_feat, dyn_feat, wih_b, whh_b, win_p, bias, out);
}
